// Decoder_33492154974538
// MI455X (gfx1250) — matrix operand order repaired and checked by reading, not compiled, not run on hardware
//
#include <hip/hip_runtime.h>

typedef unsigned short u16;
typedef __attribute__((ext_vector_type(16))) __bf16 v16bf;
typedef __attribute__((ext_vector_type(8)))  float  v8f;
typedef unsigned int v4u __attribute__((ext_vector_type(4)));
typedef int          v4i __attribute__((ext_vector_type(4)));
typedef int          v8i __attribute__((ext_vector_type(8)));

union FragBF { uint4 q[2]; v16bf v; };

#if __has_builtin(__builtin_amdgcn_tensor_load_to_lds)
#define HAVE_TDM 1
#else
#define HAVE_TDM 0
#endif

__device__ __forceinline__ u16 f2bf(float f) {
  unsigned u = __float_as_uint(f);
  u += 0x7FFFu + ((u >> 16) & 1u);   // round-to-nearest-even
  return (u16)(u >> 16);
}

__device__ __forceinline__ v8f v8f_zero() {
  v8f z = {0.f, 0.f, 0.f, 0.f, 0.f, 0.f, 0.f, 0.f};
  return z;
}

#if HAVE_TDM
// ---------------------------------------------------------------------------
// Tensor Data Mover: 2D tile load (rows x 32 bf16) global -> LDS.
// LDS padding: every 64B row, +16B pad -> 80B pitch (40 halfwords), which
// keeps every fragment b128 load 16B-aligned.
// D# layout per CDNA5 ISA.
// ---------------------------------------------------------------------------
__device__ __forceinline__ void tdm_load_tile_2d(
    const u16* gptr, unsigned ldsOff,
    unsigned tensorD0, unsigned tensorD1,   // row length (elems), #rows
    unsigned tileD0, unsigned tileD1,       // 32, 128
    unsigned strideElems) {                 // row stride in elements
  unsigned long long ga = (unsigned long long)(size_t)gptr;
  v4u g0;
  g0.x = 1u;                                              // count=1, user mode
  g0.y = ldsOff;                                          // LDS byte address
  g0.z = (unsigned)(ga & 0xFFFFFFFFull);                  // global addr lo
  g0.w = (unsigned)((ga >> 32) & 0x01FFFFFFull) | (2u << 30);  // hi + type=2
  v8i g1;
  // data_size=2B (code 1); pad_enable; pad_interval=16 DW (code 3, 64B);
  // pad_amount=4 DW (code 3, 16B)
  g1[0] = (int)((1u << 16) | (1u << 20) | (3u << 22) | (3u << 25));
  g1[1] = (int)((tensorD0 & 0xFFFFu) << 16);
  g1[2] = (int)((tensorD0 >> 16) | ((tensorD1 & 0xFFFFu) << 16));
  g1[3] = (int)((tensorD1 >> 16) | (tileD0 << 16));
  g1[4] = (int)(tileD1 & 0xFFFFu);                        // tile_dim1, dim2=0
  g1[5] = (int)strideElems;                               // dim0 stride lo32
  g1[6] = 0;
  g1[7] = 0;
  v4i z4 = {0, 0, 0, 0};
#if defined(__clang_major__) && (__clang_major__ >= 23)
  v8i z8 = {0, 0, 0, 0, 0, 0, 0, 0};
  __builtin_amdgcn_tensor_load_to_lds(g0, g1, z4, z4, z8, 0);
#else
  __builtin_amdgcn_tensor_load_to_lds(g0, g1, z4, z4, 0);
#endif
}
#endif  // HAVE_TDM

// ---------------------------------------------------------------------------
// f32 -> bf16 conversion (plain, and transposing for weight matrices)
// ---------------------------------------------------------------------------
__global__ __launch_bounds__(256) void cvt_f32_bf16_kernel(
    const float* __restrict__ in, u16* __restrict__ out, int n) {
  int i = blockIdx.x * 256 + threadIdx.x;
  if (i < n) out[i] = f2bf(in[i]);
}

// in: [K][N] row-major f32  ->  out: [N][K] row-major bf16
__global__ __launch_bounds__(256) void cvt_f32_bf16_tr_kernel(
    const float* __restrict__ in, u16* __restrict__ out, int K, int N) {
  int i = blockIdx.x * 256 + threadIdx.x;
  if (i < K * N) {
    int k = i / N, n = i - k * N;
    out[(size_t)n * K + k] = f2bf(in[i]);
  }
}

// ---------------------------------------------------------------------------
// Tiled bf16 WMMA GEMM:  C[M,N] = A[M,K] @ Bt[N,K]^T + bias (+ resid)
// A row-major bf16, Bt row-major bf16 (pre-transposed weights).
// Block tile 128x128, 8 waves (4x2), wave tile 32x64 = 2x4 WMMA accumulators.
// Double-buffered LDS tiles fed by the Tensor Data Mover (pad feature gives
// the 80B pitch); s_wait_tensorcnt fences the DMA before each barrier.
// ---------------------------------------------------------------------------
#define GEMM_BM 128
#define GEMM_BN 128
#define GEMM_BK 32
#define LDS_PITCH 40   // halfwords per row (64B data + 16B pad)

__global__ __launch_bounds__(256) void gemm_bf16_wmma_kernel(
    const u16* __restrict__ A, const u16* __restrict__ Bt,
    const float* __restrict__ bias, const float* __restrict__ resid,
    float* __restrict__ Cf, u16* __restrict__ Cb,
    int M, int N, int K) {
  __shared__ __align__(16) u16 Alds[2][GEMM_BM * LDS_PITCH];
  __shared__ __align__(16) u16 Blds[2][GEMM_BN * LDS_PITCH];

  const int tid  = threadIdx.x;
  const int lane = tid & 31;
  const int wid  = tid >> 5;
  const int g    = lane >> 4;   // lane group (0/1)
  const int ln   = lane & 15;

  const int mBase = blockIdx.y * GEMM_BM;
  const int nBase = blockIdx.x * GEMM_BN;
  const int wm = wid >> 1;      // 0..3  (rows of 32)
  const int wn = wid & 1;       // 0..1  (cols of 64)

  v8f acc[2][4];
#pragma unroll
  for (int i = 0; i < 2; ++i)
#pragma unroll
    for (int j = 0; j < 4; ++j) acc[i][j] = v8f_zero();

  const int nk = K / GEMM_BK;

#if HAVE_TDM
  // prologue: DMA tile 0 into buffer 0 (wave0 -> A, wave1 -> B)
  if (wid == 0)
    tdm_load_tile_2d(&A[(size_t)mBase * K], (unsigned)(size_t)&Alds[0][0],
                     (unsigned)K, (unsigned)M, GEMM_BK, GEMM_BM, (unsigned)K);
  if (wid == 1)
    tdm_load_tile_2d(&Bt[(size_t)nBase * K], (unsigned)(size_t)&Blds[0][0],
                     (unsigned)K, (unsigned)N, GEMM_BK, GEMM_BN, (unsigned)K);
  __builtin_amdgcn_s_wait_tensorcnt(0);
  __syncthreads();
#else
  // manual staging geometry: 128x32 tile, 16 elems (2 x b128) per thread
  const int sRow = tid >> 1;            // 0..127
  const int sCol = (tid & 1) * 16;      // 0 or 16
  {
    const u16* ag = &A[(size_t)(mBase + sRow) * K + sCol];
    const u16* bg = &Bt[(size_t)(nBase + sRow) * K + sCol];
    *(uint4*)&Alds[0][sRow * LDS_PITCH + sCol]     = *(const uint4*)ag;
    *(uint4*)&Alds[0][sRow * LDS_PITCH + sCol + 8] = *(const uint4*)(ag + 8);
    *(uint4*)&Blds[0][sRow * LDS_PITCH + sCol]     = *(const uint4*)bg;
    *(uint4*)&Blds[0][sRow * LDS_PITCH + sCol + 8] = *(const uint4*)(bg + 8);
  }
  __syncthreads();
#endif

  int cur = 0;
  for (int kk = 0; kk < nk; ++kk) {
    const int nxt = cur ^ 1;
    if (kk + 1 < nk) {
      const int kOff = (kk + 1) * GEMM_BK;
#if HAVE_TDM
      if (wid == 0)
        tdm_load_tile_2d(&A[(size_t)mBase * K + kOff],
                         (unsigned)(size_t)&Alds[nxt][0],
                         (unsigned)K, (unsigned)M, GEMM_BK, GEMM_BM, (unsigned)K);
      if (wid == 1)
        tdm_load_tile_2d(&Bt[(size_t)nBase * K + kOff],
                         (unsigned)(size_t)&Blds[nxt][0],
                         (unsigned)K, (unsigned)N, GEMM_BK, GEMM_BN, (unsigned)K);
#else
      const u16* ag = &A[(size_t)(mBase + sRow) * K + kOff + sCol];
      const u16* bg = &Bt[(size_t)(nBase + sRow) * K + kOff + sCol];
      *(uint4*)&Alds[nxt][sRow * LDS_PITCH + sCol]     = *(const uint4*)ag;
      *(uint4*)&Alds[nxt][sRow * LDS_PITCH + sCol + 8] = *(const uint4*)(ag + 8);
      *(uint4*)&Blds[nxt][sRow * LDS_PITCH + sCol]     = *(const uint4*)bg;
      *(uint4*)&Blds[nxt][sRow * LDS_PITCH + sCol + 8] = *(const uint4*)(bg + 8);
#endif
    }

    // A fragments (16x32): per-lane K chunks at g*8 and g*8+16
    const u16* Ab = &Alds[cur][0];
    const u16* Bb = &Blds[cur][0];
    FragBF afr[2];
#pragma unroll
    for (int ti = 0; ti < 2; ++ti) {
      const u16* ar = &Ab[(wm * 32 + ti * 16 + ln) * LDS_PITCH + g * 8];
      afr[ti].q[0] = *(const uint4*)ar;
      afr[ti].q[1] = *(const uint4*)(ar + 16);
    }
    // B fragments (32x16): per-lane K chunks at g*8 and g*8+16
#pragma unroll
    for (int tj = 0; tj < 4; ++tj) {
      FragBF bfr;
      const u16* br = &Bb[(wn * 64 + tj * 16 + ln) * LDS_PITCH + g * 8];
      bfr.q[0] = *(const uint4*)br;
      bfr.q[1] = *(const uint4*)(br + 16);
#pragma unroll
      for (int ti = 0; ti < 2; ++ti)
        acc[ti][tj] = __builtin_amdgcn_wmma_f32_16x16x32_bf16(
            false, afr[ti].v, false, bfr.v, (short)0, acc[ti][tj],
            false, false);
    }

#if HAVE_TDM
    __builtin_amdgcn_s_wait_tensorcnt(0);
#endif
    __syncthreads();
    cur = nxt;
  }

  // epilogue: bias (+ residual), write f32 and/or bf16
#pragma unroll
  for (int ti = 0; ti < 2; ++ti) {
#pragma unroll
    for (int tj = 0; tj < 4; ++tj) {
      const int row0 = mBase + wm * 32 + ti * 16 + g * 8;
      const int col  = nBase + wn * 64 + tj * 16 + ln;
      const float bsv = bias ? bias[col] : 0.0f;
#pragma unroll
      for (int r = 0; r < 8; ++r) {
        float v = acc[ti][tj][r] + bsv;
        const size_t idx = (size_t)(row0 + r) * N + col;
        if (resid) v += resid[idx];
        if (Cf) Cf[idx] = v;
        if (Cb) Cb[idx] = f2bf(v);
      }
    }
  }
}

// ---------------------------------------------------------------------------
// Flash-style multi-head attention (no causal mask -- reference omits it).
// One block = (batch b, head h, 128 queries). 8 waves, 16 queries per wave.
// HD = 64. Key/value tiles of 32 staged in LDS; online softmax; all matmuls
// via v_wmma_f32_16x16x32_bf16.
// ---------------------------------------------------------------------------
__global__ __launch_bounds__(256) void flash_attn_wmma_kernel(
    const u16* __restrict__ Q, int qStride,
    const u16* __restrict__ Kp, int kStride,
    const u16* __restrict__ Vp, int vStride,
    u16* __restrict__ O, int oStride, int T) {
  __shared__ __align__(16) u16 Klds[32 * 64];        // [key][d]
  __shared__ __align__(16) u16 Vt[64 * 40];          // [d][key] (+pad)
  __shared__ __align__(16) u16 Plds[8][16 * 40];     // per-wave P [q][key]

  const int tid  = threadIdx.x;
  const int lane = tid & 31;
  const int wid  = tid >> 5;
  const int g    = lane >> 4;
  const int ln   = lane & 15;

  const int h  = blockIdx.y;
  const int bb = blockIdx.z;
  const int qBase = blockIdx.x * 128;

  const u16* qhead = Q  + (size_t)bb * T * qStride + h * 64;
  const u16* khead = Kp + (size_t)bb * T * kStride + h * 64;
  const u16* vhead = Vp + (size_t)bb * T * vStride + h * 64;
  u16*       ohead = O  + (size_t)bb * T * oStride + h * 64;

  // Q fragments for this wave's 16 queries (A layout, K-dim = HD = 64)
  FragBF aq[2];
  {
    const u16* qp = &qhead[(size_t)(qBase + wid * 16 + ln) * qStride];
#pragma unroll
    for (int f = 0; f < 2; ++f) {
      aq[f].q[0] = *(const uint4*)&qp[f * 32 + g * 8];
      aq[f].q[1] = *(const uint4*)&qp[f * 32 + g * 8 + 16];
    }
  }

  v8f accO[4];
#pragma unroll
  for (int n = 0; n < 4; ++n) accO[n] = v8f_zero();
  float mrow[8], lrow[8];
#pragma unroll
  for (int r = 0; r < 8; ++r) { mrow[r] = -3.0e38f; lrow[r] = 0.0f; }

  const int kRowL = tid >> 3;        // 0..31 (key within tile)
  const int kColL = (tid & 7) * 8;   // 0..56 (d chunk)

  for (int s0 = 0; s0 < T; s0 += 32) {
    // stage K [key][d] and V transposed [d][key]
    {
      uint4 kv4 = *(const uint4*)&khead[(size_t)(s0 + kRowL) * kStride + kColL];
      *(uint4*)&Klds[kRowL * 64 + kColL] = kv4;
      uint4 vv4 = *(const uint4*)&vhead[(size_t)(s0 + kRowL) * vStride + kColL];
      u16 tmp[8];
      *(uint4*)tmp = vv4;
#pragma unroll
      for (int i = 0; i < 8; ++i)
        Vt[(kColL + i) * 40 + kRowL] = tmp[i];
    }
    __syncthreads();

    // S = Q @ K^T : two 16x16 key sub-tiles, K-dim 64 = 2 chained wmma each
    v8f sc[2];
#pragma unroll
    for (int t = 0; t < 2; ++t) {
      sc[t] = v8f_zero();
#pragma unroll
      for (int f = 0; f < 2; ++f) {
        FragBF bk;
        const u16* kr = &Klds[(t * 16 + ln) * 64 + f * 32 + g * 8];
        bk.q[0] = *(const uint4*)kr;
        bk.q[1] = *(const uint4*)(kr + 16);
        sc[t] = __builtin_amdgcn_wmma_f32_16x16x32_bf16(
            false, aq[f].v, false, bk.v, (short)0, sc[t], false, false);
      }
    }

    // online softmax update (row reductions across the 16-lane group)
    float corr[8];
#pragma unroll
    for (int r = 0; r < 8; ++r) {
      float a0 = sc[0][r] * 0.125f;   // 1/sqrt(64)
      float a1 = sc[1][r] * 0.125f;
      float mx = fmaxf(a0, a1);
      for (int d = 1; d < 16; d <<= 1) mx = fmaxf(mx, __shfl_xor(mx, d, 16));
      float mn = fmaxf(mrow[r], mx);
      corr[r] = __expf(mrow[r] - mn);
      mrow[r] = mn;
      float p0 = __expf(a0 - mn);
      float p1 = __expf(a1 - mn);
      sc[0][r] = p0;
      sc[1][r] = p1;
      float rs = p0 + p1;
      for (int d = 1; d < 16; d <<= 1) rs += __shfl_xor(rs, d, 16);
      lrow[r] = lrow[r] * corr[r] + rs;
    }
#pragma unroll
    for (int n = 0; n < 4; ++n)
#pragma unroll
      for (int r = 0; r < 8; ++r) accO[n][r] *= corr[r];

    // restage P (C layout -> A layout) through this wave's LDS slab
    {
      u16* pw = &Plds[wid][0];
#pragma unroll
      for (int t = 0; t < 2; ++t)
#pragma unroll
        for (int r = 0; r < 8; ++r)
          pw[(g * 8 + r) * 40 + t * 16 + ln] = f2bf(sc[t][r]);
    }
    __syncthreads();

    // O += P(16x32) @ V(32x64) : 4 output tiles along HD
    {
      const u16* pw = &Plds[wid][0];
      FragBF ap;
      ap.q[0] = *(const uint4*)&pw[ln * 40 + g * 8];
      ap.q[1] = *(const uint4*)&pw[ln * 40 + g * 8 + 16];
#pragma unroll
      for (int n = 0; n < 4; ++n) {
        FragBF bv;
        const u16* vr = &Vt[(n * 16 + ln) * 40 + g * 8];
        bv.q[0] = *(const uint4*)vr;
        bv.q[1] = *(const uint4*)(vr + 16);
        accO[n] = __builtin_amdgcn_wmma_f32_16x16x32_bf16(
            false, ap.v, false, bv.v, (short)0, accO[n], false, false);
      }
    }
    __syncthreads();
  }

  // normalize and store (bf16, merged [B,T,C] layout)
#pragma unroll
  for (int r = 0; r < 8; ++r) {
    const float inv = 1.0f / lrow[r];
    const size_t row = (size_t)(qBase + wid * 16 + g * 8 + r);
#pragma unroll
    for (int n = 0; n < 4; ++n)
      ohead[row * oStride + n * 16 + ln] = f2bf(accO[n][r] * inv);
  }
}

// ---------------------------------------------------------------------------
// LayerNorm over C=1024: one block per row. f32 in, f32 + optional bf16 out.
// ---------------------------------------------------------------------------
__global__ __launch_bounds__(256) void layernorm_kernel(
    const float* __restrict__ in, const float* __restrict__ w,
    const float* __restrict__ bb, float* __restrict__ outF,
    u16* __restrict__ outB) {
  __shared__ float sred[18];
  const int row = blockIdx.x;
  const int c = threadIdx.x * 4;
  const float* x = in + (size_t)row * 1024;

  float4 xv = *(const float4*)&x[c];
  float s  = xv.x + xv.y + xv.z + xv.w;
  float ss = xv.x * xv.x + xv.y * xv.y + xv.z * xv.z + xv.w * xv.w;
  for (int d = 1; d < 32; d <<= 1) {
    s  += __shfl_xor(s, d, 32);
    ss += __shfl_xor(ss, d, 32);
  }
  const int wv_id = threadIdx.x >> 5;
  if ((threadIdx.x & 31) == 0) { sred[wv_id] = s; sred[8 + wv_id] = ss; }
  __syncthreads();
  if (threadIdx.x == 0) {
    float ts = 0.f, tss = 0.f;
    for (int i = 0; i < 8; ++i) { ts += sred[i]; tss += sred[8 + i]; }
    const float mu  = ts * (1.0f / 1024.0f);
    const float var = tss * (1.0f / 1024.0f) - mu * mu;
    sred[16] = mu;
    sred[17] = rsqrtf(var + 1e-5f);
  }
  __syncthreads();
  const float mu = sred[16], rs = sred[17];

  float4 wv = *(const float4*)&w[c];
  float4 bv = *(const float4*)&bb[c];
  float o0 = (xv.x - mu) * rs * wv.x + bv.x;
  float o1 = (xv.y - mu) * rs * wv.y + bv.y;
  float o2 = (xv.z - mu) * rs * wv.z + bv.z;
  float o3 = (xv.w - mu) * rs * wv.w + bv.w;
  if (outF) {
    float4 ov = {o0, o1, o2, o3};
    *(float4*)&outF[(size_t)row * 1024 + c] = ov;
  }
  if (outB) {
    u16* ob = &outB[(size_t)row * 1024 + c];
    ob[0] = f2bf(o0); ob[1] = f2bf(o1); ob[2] = f2bf(o2); ob[3] = f2bf(o3);
  }
}

// ---------------------------------------------------------------------------
// host-side orchestration
// ---------------------------------------------------------------------------
static inline void launch_cvt(const float* in, u16* out, int n, hipStream_t s) {
  cvt_f32_bf16_kernel<<<(n + 255) / 256, 256, 0, s>>>(in, out, n);
}

static inline void launch_cvt_tr(const float* in, u16* out, int K, int N,
                                 hipStream_t s) {
  cvt_f32_bf16_tr_kernel<<<(K * N + 255) / 256, 256, 0, s>>>(in, out, K, N);
}

static inline void launch_gemm(const u16* A, const u16* Bt, const float* bias,
                               const float* resid, float* Cf, u16* Cb,
                               int M, int N, int K, hipStream_t s) {
  dim3 grid(N / GEMM_BN, M / GEMM_BM);
  gemm_bf16_wmma_kernel<<<grid, 256, 0, s>>>(A, Bt, bias, resid, Cf, Cb, M, N, K);
}

extern "C" void kernel_launch(void* const* d_in, const int* in_sizes, int n_in,
                              void* d_out, int out_size, void* d_ws, size_t ws_size,
                              hipStream_t stream) {
  (void)in_sizes; (void)n_in; (void)out_size; (void)ws_size;
  const int Bb = 4, T = 1024, C = 1024, H = 16;
  const int M = Bb * T;          // 4096 rows
  const int DFF = 4096;

  const float* x      = (const float*)d_in[0];
  const float* y      = (const float*)d_in[1];
  const float* W_attn = (const float*)d_in[2];
  const float* b_attn = (const float*)d_in[3];
  const float* W_proj = (const float*)d_in[4];
  const float* b_proj = (const float*)d_in[5];
  const float* ln_w   = (const float*)d_in[6];
  const float* ln_b   = (const float*)d_in[7];
  const float* W_en   = (const float*)d_in[8];
  const float* b_en   = (const float*)d_in[9];
  const float* W_q    = (const float*)d_in[10];
  const float* b_q    = (const float*)d_in[11];
  const float* W_cproj= (const float*)d_in[12];
  const float* b_cproj= (const float*)d_in[13];
  const float* ln1_w  = (const float*)d_in[14];
  const float* ln1_b  = (const float*)d_in[15];
  const float* ln2_w  = (const float*)d_in[16];
  const float* ln2_b  = (const float*)d_in[17];
  const float* W_d1   = (const float*)d_in[18];
  const float* b_d1   = (const float*)d_in[19];
  const float* W_d2   = (const float*)d_in[20];
  const float* b_d2   = (const float*)d_in[21];
  float* out = (float*)d_out;

  // bump allocator over workspace
  char* ws = (char*)d_ws;
  size_t off = 0;
  auto aU = [&](size_t n) { u16* p = (u16*)(ws + off); off += n * sizeof(u16); return p; };
  auto aF = [&](size_t n) { float* p = (float*)(ws + off); off += n * sizeof(float); return p; };

  const size_t MC = (size_t)M * C;
  u16* Wt_attn  = aU((size_t)C * 3 * C);   // [3C][C]
  u16* Wt_proj  = aU((size_t)C * C);       // [C][C]
  u16* Wt_en    = aU((size_t)C * 2 * C);   // [2C][C]
  u16* Wt_q     = aU((size_t)C * C);
  u16* Wt_cproj = aU((size_t)C * C);
  u16* Wt_d1    = aU((size_t)C * DFF);     // [DFF][C]
  u16* Wt_d2    = aU((size_t)DFF * C);     // [C][DFF]
  u16* y_bf   = aU(MC);
  u16* x_bf   = aU(MC);
  u16* qkv_bf = aU(MC * 3);
  u16* o_bf   = aU(MC);
  u16* kv_bf  = aU(MC * 2);
  u16* q2_bf  = aU(MC);
  u16* o2_bf  = aU(MC);
  u16* yln_bf = aU(MC);
  u16* xin_bf = aU(MC);
  u16* h_bf   = aU((size_t)M * DFF);
  float* t1    = aF(MC);
  float* yln_f = aF(MC);
  float* xin_f = aF(MC);

  // --- precision conversion; weights are transposed to [N][K] for the TDM ---
  launch_cvt_tr(W_attn,  Wt_attn,  C, 3 * C, stream);
  launch_cvt_tr(W_proj,  Wt_proj,  C, C,     stream);
  launch_cvt_tr(W_en,    Wt_en,    C, 2 * C, stream);
  launch_cvt_tr(W_q,     Wt_q,     C, C,     stream);
  launch_cvt_tr(W_cproj, Wt_cproj, C, C,     stream);
  launch_cvt_tr(W_d1,    Wt_d1,    C, DFF,   stream);
  launch_cvt_tr(W_d2,    Wt_d2,    DFF, C,   stream);
  launch_cvt(y, y_bf, (int)MC, stream);
  launch_cvt(x, x_bf, (int)MC, stream);

  dim3 attnGrid(T / 128, H, Bb);

  // --- self-attention block ---
  launch_gemm(y_bf, Wt_attn, b_attn, nullptr, nullptr, qkv_bf, M, 3 * C, C, stream);
  flash_attn_wmma_kernel<<<attnGrid, 256, 0, stream>>>(
      qkv_bf + 0, 3 * C, qkv_bf + C, 3 * C, qkv_bf + 2 * C, 3 * C,
      o_bf, C, T);
  launch_gemm(o_bf, Wt_proj, b_proj, y, t1, nullptr, M, C, C, stream);
  layernorm_kernel<<<M, 256, 0, stream>>>(t1, ln_w, ln_b, yln_f, yln_bf);

  // --- cross-attention block ---
  launch_gemm(x_bf, Wt_en, b_en, nullptr, nullptr, kv_bf, M, 2 * C, C, stream);
  launch_gemm(yln_bf, Wt_q, b_q, nullptr, nullptr, q2_bf, M, C, C, stream);
  flash_attn_wmma_kernel<<<attnGrid, 256, 0, stream>>>(
      q2_bf, C, kv_bf + 0, 2 * C, kv_bf + C, 2 * C,
      o2_bf, C, T);
  launch_gemm(o2_bf, Wt_cproj, b_cproj, yln_f, t1, nullptr, M, C, C, stream);
  layernorm_kernel<<<M, 256, 0, stream>>>(t1, ln1_w, ln1_b, xin_f, xin_bf);

  // --- feed-forward block ---
  launch_gemm(xin_bf, Wt_d1, b_d1, nullptr, nullptr, h_bf, M, DFF, C, stream);
  launch_gemm(h_bf, Wt_d2, b_d2, xin_f, t1, nullptr, M, C, DFF, stream);
  layernorm_kernel<<<M, 256, 0, stream>>>(t1, ln2_w, ln2_b, out, nullptr);
}
